// stgat_46505905881385
// MI455X (gfx1250) — matrix operand order repaired and checked by reading, not compiled, not run on hardware
//
#include <hip/hip_runtime.h>

// Problem constants (match reference)
#define BB   8
#define NN   512
#define CC   40
#define SKC  320
#define ECC  640
#define HH   8
#define OUTC 12

typedef __attribute__((ext_vector_type(16))) _Float16 v16h;
typedef __attribute__((ext_vector_type(8)))  _Float16 v8h;
typedef __attribute__((ext_vector_type(8)))  float    v8f;

// ---------------------------------------------------------------- utilities
__global__ void zero_f32(float* __restrict__ p, int n) {
  int i = blockIdx.x * blockDim.x + threadIdx.x;
  if (i < n) p[i] = 0.0f;
}

__global__ void cast_f32_to_f16(const float* __restrict__ s, _Float16* __restrict__ d, int n) {
  int i = blockIdx.x * blockDim.x + threadIdx.x;
  if (i < n) d[i] = (_Float16)s[i];
}

// ---------------------------------------------------------------- start conv
// h[b,c,n,t] = start_w[c]*x[b,0,n,t] + start_b[c] + leaky01(cat_w[c]*x[b,1,n,t] + cat_b[c])
__global__ void start_kernel(const float* __restrict__ x,
                             const float* __restrict__ sw, const float* __restrict__ sb,
                             const float* __restrict__ cw, const float* __restrict__ cb,
                             float* __restrict__ h) {
  int idx = blockIdx.x * blockDim.x + threadIdx.x;
  int total = BB * CC * NN * 13;
  if (idx >= total) return;
  int t = idx % 13; int r = idx / 13;
  int n = r % NN;   r /= NN;
  int c = r % CC;   int b = r / CC;
  float x0 = x[(((size_t)b * 2 + 0) * NN + n) * 13 + t];
  float x1 = x[(((size_t)b * 2 + 1) * NN + n) * 13 + t];
  float a = sw[c] * x0 + sb[c];
  float v = cw[c] * x1 + cb[c];
  v = v > 0.0f ? v : 0.01f * v;
  h[idx] = a + v;
}

// ---------------------------------------------------------------- gated dilated conv
__global__ void gated_conv(const float* __restrict__ cur,
                           const float* __restrict__ fw, const float* __restrict__ fb,
                           const float* __restrict__ gw, const float* __restrict__ gb,
                           float* __restrict__ gh, int t_in, int t_out, int dil) {
  int idx = blockIdx.x * blockDim.x + threadIdx.x;
  int total = BB * CC * NN * t_out;
  if (idx >= total) return;
  int tt = idx % t_out; int r = idx / t_out;
  int n = r % NN;       r /= NN;
  int c = r % CC;       int b = r / CC;
  float f = fb[c], g = gb[c];
  const float* base = cur + (size_t)b * CC * NN * t_in + (size_t)n * t_in + tt;
  for (int ci = 0; ci < CC; ++ci) {
    float v0 = base[(size_t)ci * NN * t_in];
    float v1 = base[(size_t)ci * NN * t_in + dil];
    f += fw[(c * CC + ci) * 2] * v0 + fw[(c * CC + ci) * 2 + 1] * v1;
    g += gw[(c * CC + ci) * 2] * v0 + gw[(c * CC + ci) * 2 + 1] * v1;
  }
  gh[idx] = tanhf(f) * (1.0f / (1.0f + expf(-g)));
}

// ---------------------------------------------------------------- skip 1x1 conv, tail-aligned accumulate
__global__ void skip_conv(const float* __restrict__ gh,
                          const float* __restrict__ sw, const float* __restrict__ sb,
                          float* __restrict__ skipacc, int t_out) {
  int idx = blockIdx.x * blockDim.x + threadIdx.x;
  int total = BB * SKC * NN * t_out;
  if (idx >= total) return;
  int tt = idx % t_out; int r = idx / t_out;
  int n = r % NN;       r /= NN;
  int s = r % SKC;      int b = r / SKC;
  float acc = sb[s];
  const float* gbase = gh + (size_t)b * CC * NN * t_out + (size_t)n * t_out + tt;
  for (int c = 0; c < CC; ++c) acc += sw[s * CC + c] * gbase[(size_t)c * NN * t_out];
  skipacc[(((size_t)b * SKC + s) * NN + n) * 12 + (12 - t_out) + tt] += acc;
}

// ---------------------------------------------------------------- GEMM operand packing
// A[m, k] (f16, row-major, K padded to mult of 32) from gh [B,C,N,t] : k = c*t+tt
__global__ void build_a_from_gh(const float* __restrict__ gh, _Float16* __restrict__ A,
                                int t, int Kpad) {
  int idx = blockIdx.x * blockDim.x + threadIdx.x;
  int total = BB * NN * Kpad;
  if (idx >= total) return;
  int k = idx % Kpad; int m = idx / Kpad;
  int b = m / NN, n = m % NN;
  float v = 0.0f;
  if (k < CC * t) {
    int c = k / t, tt = k % t;
    v = gh[(((size_t)b * CC + c) * NN + n) * t + tt];
  }
  A[idx] = (_Float16)v;
}

__global__ void build_a_from_f32(const float* __restrict__ src, _Float16* __restrict__ A,
                                 int F, int Kpad) {
  int idx = blockIdx.x * blockDim.x + threadIdx.x;
  int total = BB * NN * Kpad;
  if (idx >= total) return;
  int k = idx % Kpad; int m = idx / Kpad;
  A[idx] = (k < F) ? (_Float16)src[(size_t)m * F + k] : (_Float16)0.0f;
}

// Bt[n, k] (f16, K-contiguous) from W [F, H, F]:  Wmat[k][n] = W[k*Nc + n]
__global__ void build_bt(const float* __restrict__ W, _Float16* __restrict__ Bt,
                         int F, int Nc, int Kpad) {
  int idx = blockIdx.x * blockDim.x + threadIdx.x;
  int total = Nc * Kpad;
  if (idx >= total) return;
  int k = idx % Kpad; int n = idx / Kpad;
  Bt[idx] = (k < F) ? (_Float16)W[(size_t)k * Nc + n] : (_Float16)0.0f;
}

// ---------------------------------------------------------------- WMMA fragment loads
// A fragment (16x32 f16): row = lane&15; halves 0..7 -> K=8*hi.., 8..15 -> K=16+8*hi..
__device__ __forceinline__ v16h load_a_frag(const _Float16* __restrict__ p) {
  v8h l = *(const v8h*)p;
  v8h h = *(const v8h*)(p + 16);
  v16h r;
#pragma unroll
  for (int j = 0; j < 8; ++j) { r[j] = l[j]; r[8 + j] = h[j]; }
  return r;
}
// B fragment (32x16 f16 via Bt[Nc][Kpad]): col = lane&15; halves 0..7 -> K=8*hi.., 8..15 -> K=16+8*hi..
__device__ __forceinline__ v16h load_b_frag(const _Float16* __restrict__ p) {
  v8h l = *(const v8h*)p;
  v8h h = *(const v8h*)(p + 16);
  v16h r;
#pragma unroll
  for (int j = 0; j < 8; ++j) { r[j] = l[j]; r[8 + j] = h[j]; }
  return r;
}

#define WMMA_F16(a, b, c) \
  __builtin_amdgcn_wmma_f32_16x16x32_f16(false, (a), false, (b), (short)0, (c), false, false)

// ---------------------------------------------------------------- WMMA GEMM
// C[M,Nc] = A[M,Kpad](f16) x B[Kpad,Nc](f16 via Bt[Nc,Kpad]), f32 accumulate.
// One 32x64 macro-tile (2x4 WMMA tiles) per wave: 8 wmma per 6KB loaded -> ~22 FLOP/B.
// Requires M % 32 == 0, Nc % 64 == 0, Kpad % 32 == 0 (all hold for this model).
__global__ void __launch_bounds__(128) wmma_gemm(const _Float16* __restrict__ A,
                                                 const _Float16* __restrict__ Bt,
                                                 float* __restrict__ outF,
                                                 _Float16* __restrict__ outH,
                                                 int M, int Nc, int Kpad) {
  int lane = threadIdx.x & 31;
  int wave = threadIdx.x >> 5;
  int tile = blockIdx.x * 4 + wave;
  int mt = M >> 5, nt = Nc >> 6;
  if (tile >= mt * nt) return;           // uniform across the wave: EXEC all-ones inside
  int tm = tile / nt, tn = tile % nt;
  int hi = lane >> 4;                     // 0 | 1
  int lo = lane & 15;
  const _Float16* Ap0 = A + (size_t)(tm * 32 + lo) * Kpad + hi * 8;
  const _Float16* Ap1 = Ap0 + (size_t)16 * Kpad;
  const _Float16* Bp0 = Bt + (size_t)(tn * 64 + lo) * Kpad + hi * 8;
  const _Float16* Bp1 = Bp0 + (size_t)16 * Kpad;
  const _Float16* Bp2 = Bp0 + (size_t)32 * Kpad;
  const _Float16* Bp3 = Bp0 + (size_t)48 * Kpad;
  v8f c00 = {}, c01 = {}, c02 = {}, c03 = {};
  v8f c10 = {}, c11 = {}, c12 = {}, c13 = {};
  for (int k0 = 0; k0 < Kpad; k0 += 32) {
    v16h a0 = load_a_frag(Ap0 + k0);
    v16h a1 = load_a_frag(Ap1 + k0);
    v16h b0 = load_b_frag(Bp0 + k0);
    v16h b1 = load_b_frag(Bp1 + k0);
    v16h b2 = load_b_frag(Bp2 + k0);
    v16h b3 = load_b_frag(Bp3 + k0);
    c00 = WMMA_F16(a0, b0, c00);
    c01 = WMMA_F16(a0, b1, c01);
    c02 = WMMA_F16(a0, b2, c02);
    c03 = WMMA_F16(a0, b3, c03);
    c10 = WMMA_F16(a1, b0, c10);
    c11 = WMMA_F16(a1, b1, c11);
    c12 = WMMA_F16(a1, b2, c12);
    c13 = WMMA_F16(a1, b3, c13);
  }
  // C/D layout: lane holds col = lo, rows (8*hi + v) within each 16x16 tile
  int rbase = tm * 32 + hi * 8;
  int cbase = tn * 64 + lo;
  v8f accs[8] = {c00, c01, c02, c03, c10, c11, c12, c13};
  if (outH) {
#pragma unroll
    for (int q = 0; q < 8; ++q) {
      int rb = rbase + (q >> 2) * 16;
      int cb = cbase + (q & 3) * 16;
#pragma unroll
      for (int v = 0; v < 8; ++v) outH[(size_t)(rb + v) * Nc + cb] = (_Float16)accs[q][v];
    }
  } else {
#pragma unroll
    for (int q = 0; q < 8; ++q) {
      int rb = rbase + (q >> 2) * 16;
      int cb = cbase + (q & 3) * 16;
#pragma unroll
      for (int v = 0; v < 8; ++v) outF[(size_t)(rb + v) * Nc + cb] = accs[q][v];
    }
  }
}

// ---------------------------------------------------------------- GAT attention pieces
__global__ void eler_kernel(const _Float16* __restrict__ Wh,
                            const float* __restrict__ al, const float* __restrict__ ar,
                            float* __restrict__ el, float* __restrict__ er, int F) {
  int idx = blockIdx.x * blockDim.x + threadIdx.x;
  int total = BB * NN * HH;
  if (idx >= total) return;
  int h = idx % HH; int r = idx / HH;          // r = b*NN + n
  const _Float16* w = Wh + ((size_t)r * HH + h) * F;
  const float* alp = al + (size_t)h * F;
  const float* arp = ar + (size_t)h * F;
  float sl = 0.0f, sr = 0.0f;
  for (int f = 0; f < F; ++f) { float v = (float)w[f]; sl += v * alp[f]; sr += v * arp[f]; }
  el[idx] = sl; er[idx] = sr;
}

// one block per (b,n): 8-edge softmax per head, aggregate, ELU, mean over heads
__global__ void gat_aggregate(const _Float16* __restrict__ Wh,
                              const float* __restrict__ el, const float* __restrict__ er,
                              const int* __restrict__ src, float* __restrict__ out,
                              int F, int EPN) {
  int b = blockIdx.x / NN, n = blockIdx.x % NN;
  __shared__ int   s_src[16];
  __shared__ float s_alpha[HH * 16];
  int tid = threadIdx.x;
  if (tid < EPN) s_src[tid] = src[n * EPN + tid];
  __syncthreads();
  if (tid < HH) {
    int h = tid;
    float erv = er[((size_t)b * NN + n) * HH + h];
    float sc[16];
    float m = -3.0e38f;
    for (int j = 0; j < EPN; ++j) {
      float s = el[((size_t)b * NN + s_src[j]) * HH + h] + erv;
      s = s > 0.0f ? s : 0.2f * s;                 // leaky_relu(0.2)
      sc[j] = s; m = fmaxf(m, s);
    }
    float d = 0.0f;
    for (int j = 0; j < EPN; ++j) { sc[j] = expf(sc[j] - m); d += sc[j]; }
    float inv = 1.0f / d;
    for (int j = 0; j < EPN; ++j) s_alpha[h * 16 + j] = sc[j] * inv;
  }
  __syncthreads();
  for (int f = tid; f < F; f += blockDim.x) {
    float acc = 0.0f;
    for (int h = 0; h < HH; ++h) {
      float s = 0.0f;
      for (int j = 0; j < EPN; ++j)
        s += s_alpha[h * 16 + j] * (float)Wh[(((size_t)b * NN + s_src[j]) * HH + h) * F + f];
      acc += (s > 0.0f) ? s : (expf(s) - 1.0f);    // elu
    }
    out[((size_t)b * NN + n) * F + f] = acc * (1.0f / HH);
  }
}

// ---------------------------------------------------------------- combine + BN
__global__ void combine_kernel(const float* __restrict__ gh, const float* __restrict__ gatout,
                               const float* __restrict__ cur, float* __restrict__ nxt,
                               int t_in, int t_out, int dil) {
  int idx = blockIdx.x * blockDim.x + threadIdx.x;
  int total = BB * CC * NN * t_out;
  if (idx >= total) return;
  int tt = idx % t_out; int r = idx / t_out;
  int n = r % NN;       r /= NN;
  int c = r % CC;       int b = r / CC;
  int F = CC * t_out;
  float go  = gatout[((size_t)b * NN + n) * F + c * t_out + tt];
  float res = cur[(((size_t)b * CC + c) * NN + n) * t_in + dil + tt];
  nxt[idx] = gh[idx] + go + res;
}

__global__ void bn_stats(const float* __restrict__ h, float* __restrict__ stats, int t) {
  int c = blockIdx.x, tid = threadIdx.x;
  int cnt = BB * NN * t;
  float s = 0.0f, s2 = 0.0f;
  for (int i = tid; i < cnt; i += blockDim.x) {
    int b = i / (NN * t); int r = i % (NN * t);
    float v = h[((size_t)b * CC + c) * NN * t + r];
    s += v; s2 += v * v;
  }
  __shared__ float sh[256], sh2[256];
  sh[tid] = s; sh2[tid] = s2; __syncthreads();
  for (int st = 128; st > 0; st >>= 1) {
    if (tid < st) { sh[tid] += sh[tid + st]; sh2[tid] += sh2[tid + st]; }
    __syncthreads();
  }
  if (tid == 0) {
    float mu  = sh[0] / cnt;
    float var = fmaxf(sh2[0] / cnt - mu * mu, 0.0f);
    stats[c * 2]     = mu;
    stats[c * 2 + 1] = 1.0f / sqrtf(var + 1e-5f);
  }
}

__global__ void bn_apply(float* __restrict__ h, const float* __restrict__ stats,
                         const float* __restrict__ g, const float* __restrict__ bta, int t) {
  int idx = blockIdx.x * blockDim.x + threadIdx.x;
  int total = BB * CC * NN * t;
  if (idx >= total) return;
  int c = (idx / (NN * t)) % CC;
  float v = h[idx];
  h[idx] = (v - stats[c * 2]) * stats[c * 2 + 1] * g[c] + bta[c];
}

// ---------------------------------------------------------------- end stage
__global__ void build_a_skip(const float* __restrict__ skipacc, _Float16* __restrict__ A) {
  int idx = blockIdx.x * blockDim.x + threadIdx.x;
  int total = BB * NN * SKC;
  if (idx >= total) return;
  int s = idx % SKC; int m = idx / SKC;
  int b = m / NN, n = m % NN;
  float v = skipacc[(((size_t)b * SKC + s) * NN + n) * 12 + 11];
  A[idx] = (_Float16)(v > 0.0f ? v : 0.0f);
}

__global__ void end2_kernel(const float* __restrict__ end1out,
                            const float* __restrict__ e1b,
                            const float* __restrict__ e2w, const float* __restrict__ e2b,
                            float* __restrict__ out) {
  int idx = blockIdx.x * blockDim.x + threadIdx.x;
  int total = BB * OUTC * NN;
  if (idx >= total) return;
  int n = idx % NN; int r = idx / NN;
  int o = r % OUTC; int b = r / OUTC;
  float acc = e2b[o];
  const float* row = end1out + ((size_t)b * NN + n) * ECC;
  for (int e = 0; e < ECC; ++e) {
    float v = row[e] + e1b[e];
    v = v > 0.0f ? v : 0.0f;
    acc += e2w[o * ECC + e] * v;
  }
  out[idx] = acc;   // [B, 12, N, 1]
}

// ================================================================ host
static inline int gridFor(int n, int blk) { return (n + blk - 1) / blk; }

extern "C" void kernel_launch(void* const* d_in, const int* in_sizes, int n_in,
                              void* d_out, int out_size, void* d_ws, size_t ws_size,
                              hipStream_t stream) {
  (void)n_in; (void)out_size; (void)ws_size;
  const float* X       = (const float*)d_in[0];
  const int*   SRC     = (const int*)d_in[1];
  const float* start_w = (const float*)d_in[3];
  const float* start_b = (const float*)d_in[4];
  const float* cat_w   = (const float*)d_in[5];
  const float* cat_b   = (const float*)d_in[6];
  const float *filt_w[8], *filt_b[8], *gate_w[8], *gate_b[8];
  const float *skip_w[8], *skip_b[8], *bn_g[8], *bn_b[8];
  for (int i = 0; i < 8; ++i) {
    filt_w[i] = (const float*)d_in[7 + i];
    filt_b[i] = (const float*)d_in[15 + i];
    gate_w[i] = (const float*)d_in[23 + i];
    gate_b[i] = (const float*)d_in[31 + i];
    skip_w[i] = (const float*)d_in[39 + i];
    skip_b[i] = (const float*)d_in[47 + i];
    bn_g[i]   = (const float*)d_in[55 + i];
    bn_b[i]   = (const float*)d_in[63 + i];
  }
  const float *gA_W[7], *gA_al[7], *gA_ar[7], *gB_W[7], *gB_al[7], *gB_ar[7];
  for (int i = 0; i < 7; ++i) {
    gA_W[i]  = (const float*)d_in[71 + 3 * i];
    gA_al[i] = (const float*)d_in[72 + 3 * i];
    gA_ar[i] = (const float*)d_in[73 + 3 * i];
    gB_W[i]  = (const float*)d_in[92 + 3 * i];
    gB_al[i] = (const float*)d_in[93 + 3 * i];
    gB_ar[i] = (const float*)d_in[94 + 3 * i];
  }
  const float* end1_w = (const float*)d_in[113];
  const float* end1_b = (const float*)d_in[114];
  const float* end2_w = (const float*)d_in[115];
  const float* end2_b = (const float*)d_in[116];

  const int E   = in_sizes[1];
  const int EPN = E / NN;                     // 8 in-edges per node (dst = repeat(arange(N)))
  const int M   = BB * NN;                    // 4096 GEMM rows

  // ---- workspace carve-up
  char* w = (char*)d_ws; size_t off = 0;
  auto give = [&](size_t bytes) -> char* {
    char* p = w + off; off += (bytes + 255) & ~(size_t)255; return p;
  };
  float*     hA      = (float*)give((size_t)BB * CC * NN * 13 * 4);
  float*     hB      = (float*)give((size_t)BB * CC * NN * 13 * 4);
  float*     gh      = (float*)give((size_t)BB * CC * NN * 12 * 4);
  float*     skipacc = (float*)give((size_t)BB * SKC * NN * 12 * 4);
  _Float16*  Af16    = (_Float16*)give((size_t)M * 480 * 2);
  _Float16*  Btf16   = (_Float16*)give((size_t)3840 * 480 * 2);
  _Float16*  Whf16   = (_Float16*)give((size_t)M * HH * 480 * 2);
  float*     el      = (float*)give((size_t)BB * NN * HH * 4);
  float*     er      = (float*)give((size_t)BB * NN * HH * 4);
  float*     gatout  = (float*)give((size_t)M * 480 * 4);
  float*     stats   = (float*)give((size_t)CC * 2 * 4);
  float*     end1out = (float*)give((size_t)M * ECC * 4);

  // ---- init
  {
    int n = BB * SKC * NN * 12;
    zero_f32<<<gridFor(n, 256), 256, 0, stream>>>(skipacc, n);
    int t = BB * CC * NN * 13;
    start_kernel<<<gridFor(t, 256), 256, 0, stream>>>(X, start_w, start_b, cat_w, cat_b, hA);
  }

  static const int TSa[8]  = {13, 12, 10, 9, 7, 6, 4, 3};
  static const int DILa[8] = {1, 2, 1, 2, 1, 2, 1, 2};
  float* cur = hA;
  float* nxt = hB;

  for (int i = 0; i < 8; ++i) {
    int t_in = TSa[i], dil = DILa[i], t_out = t_in - dil;
    {
      int n = BB * CC * NN * t_out;
      gated_conv<<<gridFor(n, 256), 256, 0, stream>>>(cur, filt_w[i], filt_b[i],
                                                      gate_w[i], gate_b[i], gh, t_in, t_out, dil);
    }
    {
      int n = BB * SKC * NN * t_out;
      skip_conv<<<gridFor(n, 256), 256, 0, stream>>>(gh, skip_w[i], skip_b[i], skipacc, t_out);
    }
    if (i == 7) break;

    int F    = CC * t_out;
    int Kpad = ((F + 31) / 32) * 32;
    int Nc   = HH * F;
    int tiles = (M / 32) * (Nc / 64);          // 32x64 macro-tiles, one per wave

    // ---- GAT A
    {
      int n = M * Kpad;
      build_a_from_gh<<<gridFor(n, 256), 256, 0, stream>>>(gh, Af16, t_out, Kpad);
      int n2 = Nc * Kpad;
      build_bt<<<gridFor(n2, 256), 256, 0, stream>>>(gA_W[i], Btf16, F, Nc, Kpad);
      wmma_gemm<<<gridFor(tiles, 4), 128, 0, stream>>>(Af16, Btf16, (float*)nullptr, Whf16, M, Nc, Kpad);
      int n3 = BB * NN * HH;
      eler_kernel<<<gridFor(n3, 256), 256, 0, stream>>>(Whf16, gA_al[i], gA_ar[i], el, er, F);
      gat_aggregate<<<BB * NN, 128, 0, stream>>>(Whf16, el, er, SRC, gatout, F, EPN);
    }
    // ---- GAT B
    {
      int n = M * Kpad;
      build_a_from_f32<<<gridFor(n, 256), 256, 0, stream>>>(gatout, Af16, F, Kpad);
      int n2 = Nc * Kpad;
      build_bt<<<gridFor(n2, 256), 256, 0, stream>>>(gB_W[i], Btf16, F, Nc, Kpad);
      wmma_gemm<<<gridFor(tiles, 4), 128, 0, stream>>>(Af16, Btf16, (float*)nullptr, Whf16, M, Nc, Kpad);
      int n3 = BB * NN * HH;
      eler_kernel<<<gridFor(n3, 256), 256, 0, stream>>>(Whf16, gB_al[i], gB_ar[i], el, er, F);
      gat_aggregate<<<BB * NN, 128, 0, stream>>>(Whf16, el, er, SRC, gatout, F, EPN);
    }
    // ---- combine + BN
    {
      int n = BB * CC * NN * t_out;
      combine_kernel<<<gridFor(n, 256), 256, 0, stream>>>(gh, gatout, cur, nxt, t_in, t_out, dil);
      bn_stats<<<CC, 256, 0, stream>>>(nxt, stats, t_out);
      bn_apply<<<gridFor(n, 256), 256, 0, stream>>>(nxt, stats, bn_g[i], bn_b[i], t_out);
    }
    float* tmp = cur; cur = nxt; nxt = tmp;
  }

  // ---- end stage: relu(skip) -> 1x1 (WMMA GEMM) -> relu -> 1x1 -> out
  {
    int n = M * SKC;
    build_a_skip<<<gridFor(n, 256), 256, 0, stream>>>(skipacc, Af16);
    int n2 = ECC * SKC;
    cast_f32_to_f16<<<gridFor(n2, 256), 256, 0, stream>>>(end1_w, Btf16, n2); // Bt[e][sk] == end1_w layout
    int tiles = (M / 32) * (ECC / 64);
    wmma_gemm<<<gridFor(tiles, 4), 128, 0, stream>>>(Af16, Btf16, end1out, (_Float16*)nullptr, M, ECC, SKC);
    int n3 = BB * OUTC * NN;
    end2_kernel<<<gridFor(n3, 256), 256, 0, stream>>>(end1out, end1_b, end2_w, end2_b, (float*)d_out);
  }
}
